// BottleneckShared_70222715289973
// MI455X (gfx1250) — matrix operand order repaired and checked by reading, not compiled, not run on hardware
//
#include <hip/hip_runtime.h>
#include <hip/hip_bf16.h>

typedef __attribute__((ext_vector_type(16))) __bf16 v16bf;
typedef __attribute__((ext_vector_type(8)))  float  v8f;

#define BN_EPS 1e-5f

__device__ __forceinline__ unsigned short f2bf(float f) {
    unsigned u = __float_as_uint(f);
    u += 0x7FFFu + ((u >> 16) & 1u);          // round-to-nearest-even
    return (unsigned short)(u >> 16);
}

union FragBF { v16bf v; uint4 q[2]; };

__device__ __forceinline__ void loadA(FragBF& f, const unsigned short* row, int k0, int h) {
    f.q[0] = *(const uint4*)(row + k0 + h * 8);
    f.q[1] = *(const uint4*)(row + k0 + 16 + h * 8);
}
__device__ __forceinline__ void loadB(FragBF& f, const unsigned short* row, int k0, int h) {
    f.q[0] = *(const uint4*)(row + k0 + h * 8);
    f.q[1] = *(const uint4*)(row + k0 + 16 + h * 8);
}
__device__ __forceinline__ v8f wmma_bf16(const FragBF& a, const FragBF& b, v8f c) {
    return __builtin_amdgcn_wmma_f32_16x16x32_bf16(false, a.v, false, b.v, (short)0, c, false, false);
}

// pack 8 BN+ReLU'd floats into 8 bf16 and store as one b128
__device__ __forceinline__ void store_bnrelu_bf16(v8f acc, const float* s, const float* sh,
                                                  int cbase, unsigned short* dst) {
    unsigned short ob[8];
    for (int r = 0; r < 8; ++r)
        ob[r] = f2bf(fmaxf(acc[r] * s[cbase + r] + sh[cbase + r], 0.f));
    uint4 pk;
    pk.x = (unsigned)ob[0] | ((unsigned)ob[1] << 16);
    pk.y = (unsigned)ob[2] | ((unsigned)ob[3] << 16);
    pk.z = (unsigned)ob[4] | ((unsigned)ob[5] << 16);
    pk.w = (unsigned)ob[6] | ((unsigned)ob[7] << 16);
    *(uint4*)(dst + cbase) = pk;
}

// ---------------- router: global average pool (one wave32 per (b,c)) ----------------
__global__ void k_pool(const float* __restrict__ x, float* __restrict__ pooled) {
    int wave = (blockIdx.x << 3) + (threadIdx.x >> 5);
    int lane = threadIdx.x & 31;
    int b = wave >> 9;
    int c = wave & 511;
    const float* p = x + ((size_t)b * 512 + c) * 784;
    float s = 0.f;
    for (int i = lane; i < 784; i += 32) s += p[i];
    for (int m = 16; m; m >>= 1) s += __shfl_xor(s, m, 32);
    if (lane == 0) pooled[b * 512 + c] = s * (1.0f / 784.0f);
}

// ---------------- router: linear + sigmoid (512 threads = 64 b x 8 e) ----------------
__global__ void k_router(const float* __restrict__ pooled, const float* __restrict__ rwgt,
                         const float* __restrict__ rb, float* __restrict__ rw) {
    int t = threadIdx.x;
    int b = t >> 3, e = t & 7;
    const float* p = pooled + b * 512;
    const float* w = rwgt + e * 512;
    float s = 0.f;
    for (int i = 0; i < 512; ++i) s += p[i] * w[i];
    s += rb[e];
    rw[b * 8 + e] = 1.0f / (1.0f + __expf(-s));
}

// ---------------- fold BN params into per-channel scale/shift ----------------
__global__ void k_bn(const float* g1, const float* b1, const float* m1, const float* v1,
                     const float* g2, const float* b2, const float* m2, const float* v2,
                     const float* g3, const float* b3, const float* m3, const float* v3,
                     float* s1, float* sh1, float* s2, float* sh2, float* s3, float* sh3) {
    int i = blockIdx.x * blockDim.x + threadIdx.x;
    if (i < 128) { float s = g1[i] * rsqrtf(v1[i] + BN_EPS); s1[i] = s; sh1[i] = b1[i] - m1[i] * s; }
    if (i < 128) { float s = g2[i] * rsqrtf(v2[i] + BN_EPS); s2[i] = s; sh2[i] = b2[i] - m2[i] * s; }
    if (i < 512) { float s = g3[i] * rsqrtf(v3[i] + BN_EPS); s3[i] = s; sh3[i] = b3[i] - m3[i] * s; }
}

// ---------------- NCHW fp32 -> [b][hw][512] bf16, LDS-tiled transpose ----------------
__global__ void k_nchw2hwc(const float* __restrict__ x, unsigned short* __restrict__ xb) {
    __shared__ float t[32][33];
    int b  = blockIdx.z;
    int c0 = blockIdx.y * 32;
    int h0 = blockIdx.x * 32;
    const float* xp = x + (size_t)b * 512 * 784;
    for (int r = threadIdx.y; r < 32; r += 8) {
        int hw = h0 + threadIdx.x;
        t[r][threadIdx.x] = (hw < 784) ? xp[(size_t)(c0 + r) * 784 + hw] : 0.f;
    }
    __syncthreads();
    unsigned short* op = xb + (size_t)b * 784 * 512;
    for (int r = threadIdx.y; r < 32; r += 8) {
        int hw = h0 + r;
        if (hw < 784) op[(size_t)hw * 512 + c0 + threadIdx.x] = f2bf(t[threadIdx.x][r]);
    }
}

// ---------------- per-sample expert mix for 1x1 weights ----------------
__global__ void k_mix11(const float* __restrict__ w, const float* __restrict__ rw,
                        unsigned short* __restrict__ out, int n) {
    int b = blockIdx.y;
    float rr[8];
    for (int e = 0; e < 8; ++e) rr[e] = rw[b * 8 + e];
    for (int i = blockIdx.x * blockDim.x + threadIdx.x; i < n; i += gridDim.x * blockDim.x) {
        float s = 0.f;
        for (int e = 0; e < 8; ++e) s += rr[e] * w[(size_t)e * n + i];
        out[(size_t)b * n + i] = f2bf(s);
    }
}

// ---------------- per-sample mix for 3x3: src [e][o][i][t] -> out [b][t][o][i] ----------
__global__ void k_mix33(const float* __restrict__ w2, const float* __restrict__ rw,
                        unsigned short* __restrict__ out) {
    const int OI = 128 * 128;
    int b = blockIdx.y;
    float rr[8];
    for (int e = 0; e < 8; ++e) rr[e] = rw[b * 8 + e];
    for (int idx = blockIdx.x * blockDim.x + threadIdx.x; idx < 9 * OI; idx += gridDim.x * blockDim.x) {
        int t  = idx / OI;
        int oi = idx - t * OI;
        int o = oi >> 7, i = oi & 127;
        size_t src = ((size_t)o * 128 + i) * 9 + t;
        float s = 0.f;
        for (int e = 0; e < 8; ++e) s += rr[e] * w2[(size_t)e * 9 * OI + src];
        out[(size_t)b * 9 * OI + idx] = f2bf(s);
    }
}

// ---------------- conv1: 1x1 GEMM, 2x2 register-blocked (32x32 out per wave) ----------
__global__ void k_conv1(const unsigned short* __restrict__ xb,  // [B][784][512]
                        const unsigned short* __restrict__ W1,  // [B][128][512]
                        const float* __restrict__ s1, const float* __restrict__ sh1,
                        unsigned short* __restrict__ act1)      // [B][784][128]
{
    int wave = (blockIdx.x << 3) + (threadIdx.x >> 5);
    int lane = threadIdx.x & 31;
    int b   = wave / 100;              // 4 Mgroups * 25 Ngroups
    int rem = wave - b * 100;
    int mg  = rem / 25;                // 0..3  -> rows mg*32
    int ng  = rem - mg * 25;           // 0..24 -> cols ng*32
    int h = lane >> 4, l16 = lane & 15;

    int m0 = mg * 32, n0 = ng * 32;
    int hw0 = n0 + l16;
    int hw1 = n0 + 16 + l16;
    int hw1c = hw1 < 784 ? hw1 : 783;  // clamp reads for the odd tail

    const unsigned short* A0 = W1 + ((size_t)b * 128 + m0 + l16) * 512;
    const unsigned short* A1 = A0 + (size_t)16 * 512;
    const unsigned short* B0 = xb + ((size_t)b * 784 + hw0) * 512;
    const unsigned short* B1 = xb + ((size_t)b * 784 + hw1c) * 512;

    v8f c00 = {}, c01 = {}, c10 = {}, c11 = {};
    for (int k0 = 0; k0 < 512; k0 += 32) {
        FragBF a0, a1, b0, b1;
        loadA(a0, A0, k0, h);  loadA(a1, A1, k0, h);
        loadB(b0, B0, k0, h);  loadB(b1, B1, k0, h);
        c00 = wmma_bf16(a0, b0, c00);
        c01 = wmma_bf16(a0, b1, c01);
        c10 = wmma_bf16(a1, b0, c10);
        c11 = wmma_bf16(a1, b1, c11);
    }
    int cb0 = m0 + h * 8, cb1 = m0 + 16 + h * 8;
    unsigned short* o0 = act1 + ((size_t)b * 784 + hw0) * 128;
    store_bnrelu_bf16(c00, s1, sh1, cb0, o0);
    store_bnrelu_bf16(c10, s1, sh1, cb1, o0);
    if (hw1 < 784) {
        unsigned short* o1 = act1 + ((size_t)b * 784 + hw1) * 128;
        store_bnrelu_bf16(c01, s1, sh1, cb0, o1);
        store_bnrelu_bf16(c11, s1, sh1, cb1, o1);
    }
}

// ---------------- conv2: 3x3 implicit GEMM, 2x2 blocked, 9 taps x K=128 ---------------
__global__ void k_conv2(const unsigned short* __restrict__ act1, // [B][784][128]
                        const unsigned short* __restrict__ W2,   // [B][9][128][128]
                        const float* __restrict__ s2, const float* __restrict__ sh2,
                        unsigned short* __restrict__ act2)       // [B][784][128]
{
    int wave = (blockIdx.x << 3) + (threadIdx.x >> 5);
    int lane = threadIdx.x & 31;
    int b   = wave / 100;
    int rem = wave - b * 100;
    int mg  = rem / 25;
    int ng  = rem - mg * 25;
    int h = lane >> 4, l16 = lane & 15;

    int m0 = mg * 32, n0 = ng * 32;
    int hw0 = n0 + l16;
    int hw1 = n0 + 16 + l16;
    int hw1c = hw1 < 784 ? hw1 : 783;
    int oy0 = hw0 / 28,  ox0 = hw0 - oy0 * 28;
    int oy1 = hw1c / 28, ox1 = hw1c - oy1 * 28;

    const unsigned short* Abase = W2 + (size_t)b * 9 * 16384 + (size_t)(m0 + l16) * 128;
    const unsigned short* actb  = act1 + (size_t)b * 784 * 128;

    v8f c00 = {}, c01 = {}, c10 = {}, c11 = {};
    for (int t = 0; t < 9; ++t) {
        int kh = t / 3, kw = t - kh * 3;
        int iy0 = oy0 + kh - 1, ix0 = ox0 + kw - 1;
        int iy1 = oy1 + kh - 1, ix1 = ox1 + kw - 1;
        bool in0 = (iy0 >= 0) & (iy0 < 28) & (ix0 >= 0) & (ix0 < 28);
        bool in1 = (iy1 >= 0) & (iy1 < 28) & (ix1 >= 0) & (ix1 < 28);
        const unsigned short* R0 = actb + (size_t)(iy0 * 28 + ix0) * 128;
        const unsigned short* R1 = actb + (size_t)(iy1 * 28 + ix1) * 128;
        const unsigned short* A0 = Abase + (size_t)t * 16384;
        const unsigned short* A1 = A0 + 16 * 128;
        for (int k0 = 0; k0 < 128; k0 += 32) {
            FragBF a0, a1, b0, b1;
            loadA(a0, A0, k0, h);  loadA(a1, A1, k0, h);
            if (in0) loadB(b0, R0, k0, h);
            else { b0.q[0] = make_uint4(0,0,0,0); b0.q[1] = make_uint4(0,0,0,0); }
            if (in1) loadB(b1, R1, k0, h);
            else { b1.q[0] = make_uint4(0,0,0,0); b1.q[1] = make_uint4(0,0,0,0); }
            c00 = wmma_bf16(a0, b0, c00);
            c01 = wmma_bf16(a0, b1, c01);
            c10 = wmma_bf16(a1, b0, c10);
            c11 = wmma_bf16(a1, b1, c11);
        }
    }
    int cb0 = m0 + h * 8, cb1 = m0 + 16 + h * 8;
    unsigned short* o0 = act2 + ((size_t)b * 784 + hw0) * 128;
    store_bnrelu_bf16(c00, s2, sh2, cb0, o0);
    store_bnrelu_bf16(c10, s2, sh2, cb1, o0);
    if (hw1 < 784) {
        unsigned short* o1 = act2 + ((size_t)b * 784 + hw1) * 128;
        store_bnrelu_bf16(c01, s2, sh2, cb0, o1);
        store_bnrelu_bf16(c11, s2, sh2, cb1, o1);
    }
}

// ---------------- conv3: 1x1 GEMM 2x2 blocked + BN + residual + ReLU -> fp32 ----------
__global__ void k_conv3(const unsigned short* __restrict__ act2, // [B][784][128]
                        const unsigned short* __restrict__ W3,   // [B][512][128]
                        const float* __restrict__ s3, const float* __restrict__ sh3,
                        const float* __restrict__ x,             // residual, NCHW fp32
                        float* __restrict__ out)                 // NCHW fp32
{
    int wave = (blockIdx.x << 3) + (threadIdx.x >> 5);
    int lane = threadIdx.x & 31;
    int b   = wave / 400;              // 16 Mgroups * 25 Ngroups
    int rem = wave - b * 400;
    int mg  = rem / 25;                // 0..15
    int ng  = rem - mg * 25;           // 0..24
    int h = lane >> 4, l16 = lane & 15;

    int m0 = mg * 32, n0 = ng * 32;
    int hw0 = n0 + l16;
    int hw1 = n0 + 16 + l16;
    int hw1c = hw1 < 784 ? hw1 : 783;

    const unsigned short* A0 = W3 + ((size_t)b * 512 + m0 + l16) * 128;
    const unsigned short* A1 = A0 + (size_t)16 * 128;
    const unsigned short* B0 = act2 + ((size_t)b * 784 + hw0) * 128;
    const unsigned short* B1 = act2 + ((size_t)b * 784 + hw1c) * 128;

    v8f c00 = {}, c01 = {}, c10 = {}, c11 = {};
    for (int k0 = 0; k0 < 128; k0 += 32) {
        FragBF a0, a1, b0, b1;
        loadA(a0, A0, k0, h);  loadA(a1, A1, k0, h);
        loadB(b0, B0, k0, h);  loadB(b1, B1, k0, h);
        c00 = wmma_bf16(a0, b0, c00);
        c01 = wmma_bf16(a0, b1, c01);
        c10 = wmma_bf16(a1, b0, c10);
        c11 = wmma_bf16(a1, b1, c11);
    }
    bool v1ok = hw1 < 784;
    for (int r = 0; r < 8; ++r) {
        int cA = m0 + h * 8 + r;
        int cB = cA + 16;
        size_t i00 = ((size_t)b * 512 + cA) * 784 + hw0;
        size_t i10 = ((size_t)b * 512 + cB) * 784 + hw0;
        out[i00] = fmaxf(c00[r] * s3[cA] + sh3[cA] + x[i00], 0.f);
        out[i10] = fmaxf(c10[r] * s3[cB] + sh3[cB] + x[i10], 0.f);
        if (v1ok) {
            size_t i01 = ((size_t)b * 512 + cA) * 784 + hw1;
            size_t i11 = ((size_t)b * 512 + cB) * 784 + hw1;
            out[i01] = fmaxf(c01[r] * s3[cA] + sh3[cA] + x[i01], 0.f);
            out[i11] = fmaxf(c11[r] * s3[cB] + sh3[cB] + x[i11], 0.f);
        }
    }
}

extern "C" void kernel_launch(void* const* d_in, const int* in_sizes, int n_in,
                              void* d_out, int out_size, void* d_ws, size_t ws_size,
                              hipStream_t stream) {
    const float* x   = (const float*)d_in[0];
    const float* rwg = (const float*)d_in[1];
    const float* rbi = (const float*)d_in[2];
    const float* w1  = (const float*)d_in[3];
    const float* w2  = (const float*)d_in[4];
    const float* w3  = (const float*)d_in[5];
    const float* g1 = (const float*)d_in[6],  *b1 = (const float*)d_in[7];
    const float* m1 = (const float*)d_in[8],  *v1 = (const float*)d_in[9];
    const float* g2 = (const float*)d_in[10], *b2 = (const float*)d_in[11];
    const float* m2 = (const float*)d_in[12], *v2 = (const float*)d_in[13];
    const float* g3 = (const float*)d_in[14], *b3 = (const float*)d_in[15];
    const float* m3 = (const float*)d_in[16], *v3 = (const float*)d_in[17];
    float* out = (float*)d_out;

    char* ws = (char*)d_ws;
    size_t off = 0;
    auto alloc = [&](size_t bytes) { void* p = ws + off; off += (bytes + 255) & ~(size_t)255; return p; };
    float* pooled = (float*)alloc(64 * 512 * 4);
    float* rw     = (float*)alloc(64 * 8 * 4);
    float* s1  = (float*)alloc(128 * 4);  float* sh1 = (float*)alloc(128 * 4);
    float* s2  = (float*)alloc(128 * 4);  float* sh2 = (float*)alloc(128 * 4);
    float* s3  = (float*)alloc(512 * 4);  float* sh3 = (float*)alloc(512 * 4);
    unsigned short* xb   = (unsigned short*)alloc((size_t)64 * 784 * 512 * 2);
    unsigned short* W1c  = (unsigned short*)alloc((size_t)64 * 128 * 512 * 2);
    unsigned short* W2c  = (unsigned short*)alloc((size_t)64 * 9 * 128 * 128 * 2);
    unsigned short* W3c  = (unsigned short*)alloc((size_t)64 * 512 * 128 * 2);
    unsigned short* act1 = (unsigned short*)alloc((size_t)64 * 784 * 128 * 2);
    unsigned short* act2 = (unsigned short*)alloc((size_t)64 * 784 * 128 * 2);
    (void)ws_size; (void)n_in; (void)in_sizes; (void)out_size;

    k_pool<<<4096, 256, 0, stream>>>(x, pooled);
    k_router<<<1, 512, 0, stream>>>(pooled, rwg, rbi, rw);
    k_bn<<<2, 256, 0, stream>>>(g1, b1, m1, v1, g2, b2, m2, v2, g3, b3, m3, v3,
                                s1, sh1, s2, sh2, s3, sh3);
    k_nchw2hwc<<<dim3(25, 16, 64), dim3(32, 8), 0, stream>>>(x, xb);
    k_mix11<<<dim3(256, 64), 256, 0, stream>>>(w1, rw, W1c, 128 * 512);
    k_mix33<<<dim3(288, 64), 256, 0, stream>>>(w2, rw, W2c);
    k_mix11<<<dim3(256, 64), 256, 0, stream>>>(w3, rw, W3c, 512 * 128);
    // 2x2 register-blocked WMMA GEMMs: 8 waves/block, 32x32 output per wave
    k_conv1<<<800, 256, 0, stream>>>(xb, W1c, s1, sh1, act1);       // 64*4*25 waves
    k_conv2<<<800, 256, 0, stream>>>(act1, W2c, s2, sh2, act2);     // 64*4*25 waves
    k_conv3<<<3200, 256, 0, stream>>>(act2, W3c, s3, sh3, x, out);  // 64*16*25 waves
}
